// Biaffine_12910671691949
// MI455X (gfx1250) — matrix operand order repaired and checked by reading, not compiled, not run on hardware
//
#include <hip/hip_runtime.h>
#include <hip/hip_bf16.h>

typedef __attribute__((ext_vector_type(16))) _Float16 v16h;
typedef __attribute__((ext_vector_type(8)))  _Float16 v8h;
typedef __attribute__((ext_vector_type(8)))  float    v8f;
typedef __attribute__((ext_vector_type(4)))  float    v4f;

#define BB   32
#define SS   255
#define HH   1024
#define BS   120
#define BSP  128
#define OO   10
#define WD   20
#define HSZ  262            // 2*(BS+1)+WD
#define MM   (BB*SS)        // 8160
#define MT   (MM/16)        // 510 (exact)
#define NU   (OO*BSP)       // 1280

// ---------------- fragment loaders (wave32 WMMA 16x16x32 f16) ----------------
// A (MxK row-major f16): lane holds row m=lane%16; halves K = [8g,8g+8) and [16+8g,16+8g+8)
__device__ __forceinline__ v16h frag_a_f16(const _Float16* __restrict__ p, int ld, int lane) {
  int m = lane & 15, g = lane >> 4;
  const _Float16* q = p + (size_t)m * ld + g * 8;
  v8h lo = *(const v8h*)(q);
  v8h hi = *(const v8h*)(q + 16);
  v16h r;
#pragma unroll
  for (int i = 0; i < 8; ++i) { r[i] = lo[i]; r[i + 8] = hi[i]; }
  return r;
}

// A from f32 source, converted in-register
__device__ __forceinline__ v16h frag_a_f32(const float* __restrict__ p, int ld, int lane) {
  int m = lane & 15, g = lane >> 4;
  const float* q = p + (size_t)m * ld + g * 8;
  v4f a0 = *(const v4f*)(q);
  v4f a1 = *(const v4f*)(q + 4);
  v4f a2 = *(const v4f*)(q + 16);
  v4f a3 = *(const v4f*)(q + 20);
  v16h r;
#pragma unroll
  for (int i = 0; i < 4; ++i) {
    r[i]      = (_Float16)a0[i];
    r[i + 4]  = (_Float16)a1[i];
    r[i + 8]  = (_Float16)a2[i];
    r[i + 12] = (_Float16)a3[i];
  }
  return r;
}

// B from Bt (NxK row-major f16, generic pointer — works for global and LDS):
// lane holds col n=lane%16; halves K = [8g,8g+8) and [16+8g,16+8g+8)
__device__ __forceinline__ v16h frag_b_f16(const _Float16* p, int ld, int lane) {
  int n = lane & 15, g = lane >> 4;
  const _Float16* q = p + (size_t)n * ld + g * 8;
  v8h lo = *(const v8h*)(q);
  v8h hi = *(const v8h*)(q + 16);
  v16h r;
#pragma unroll
  for (int i = 0; i < 8; ++i) { r[i] = lo[i]; r[i + 8] = hi[i]; }
  return r;
}

__device__ __forceinline__ v8f wmma32(v16h a, v16h b, v8f c) {
  return __builtin_amdgcn_wmma_f32_16x16x32_f16(false, a, false, b, (short)0, c, false, false);
}

// ---------------- pack kernels ----------------
__global__ void k_pack_w(const float* __restrict__ hw, const float* __restrict__ tw,
                         _Float16* __restrict__ hwt, _Float16* __restrict__ twt) {
  int tid = blockIdx.x * blockDim.x + threadIdx.x;
  if (tid >= BSP * HH) return;
  int n = tid / HH, k = tid % HH;                 // Bt[n][k] = W[k][n]
  float v0 = (n < BS) ? hw[(size_t)k * BS + n] : 0.f;
  float v1 = (n < BS) ? tw[(size_t)k * BS + n] : 0.f;
  hwt[tid] = (_Float16)v0;
  twt[tid] = (_Float16)v1;
}

__global__ void k_pack_u(const float* __restrict__ U, _Float16* __restrict__ ut) {
  int tid = blockIdx.x * blockDim.x + threadIdx.x;
  if (tid >= NU * BSP) return;
  int n = tid / BSP, i = tid % BSP;               // ut[(o*BSP+j)][i] = U[o][i][j]
  int o = n / BSP, j = n % BSP;
  float v = (i < BS && j < BS) ? U[((size_t)o * BS + i) * BS + j] : 0.f;
  ut[tid] = (_Float16)v;
}

__global__ void k_pack_wtab(const float* __restrict__ wtable, const float* __restrict__ cls_w,
                            float* __restrict__ wtab) {
  int tid = blockIdx.x * blockDim.x + threadIdx.x;
  if (tid >= 256 * OO) return;
  int d = tid / OO, o = tid % OO;
  const float* ww = cls_w + (size_t)o * HSZ + 2 * (BS + 1);
  float acc = 0.f;
#pragma unroll
  for (int w = 0; w < WD; ++w) acc += wtable[d * WD + w] * ww[w];
  wtab[tid] = acc;
}

// ---------------- GEMM1: h/t = leaky_relu(state @ W + b), f16 out ----------------
__global__ __launch_bounds__(128) void k_proj(const float* __restrict__ state,
                                              const _Float16* __restrict__ wt,
                                              const float* __restrict__ bias,
                                              _Float16* __restrict__ outh) {
  int lane = threadIdx.x & 31, wave = threadIdx.x >> 5;
  int mt = blockIdx.x;
  const float* A = state + (size_t)mt * 16 * HH;
  int n0 = wave * 32;                              // wave covers 2 N-tiles
  v8f c0 = {}, c1 = {};
  for (int kk = 0; kk < HH; kk += 32) {
    v16h a  = frag_a_f32(A + kk, HH, lane);
    v16h b0 = frag_b_f16(wt + (size_t)n0 * HH + kk, HH, lane);
    v16h b1 = frag_b_f16(wt + (size_t)(n0 + 16) * HH + kk, HH, lane);
    c0 = wmma32(a, b0, c0);
    c1 = wmma32(a, b1, c1);
  }
  int g = lane >> 4, nl = lane & 15;
#pragma unroll
  for (int v = 0; v < 8; ++v) {
    int m = mt * 16 + v + 8 * g;
    int na = n0 + nl, nb = n0 + 16 + nl;
    float xa = c0[v] + ((na < BS) ? bias[na] : 0.f);
    float xb = c1[v] + ((nb < BS) ? bias[nb] : 0.f);
    xa = (xa >= 0.f) ? xa : 0.01f * xa;
    xb = (xb >= 0.f) ? xb : 0.01f * xb;
    outh[(size_t)m * BSP + na] = (_Float16)xa;
    outh[(size_t)m * BSP + nb] = (_Float16)xb;
  }
}

// ---------------- bias terms: hb = h1·Wh + cls_b, tb = t1·Wt ----------------
__global__ void k_bias(const _Float16* __restrict__ h16, const _Float16* __restrict__ t16,
                       const float* __restrict__ cls_w, const float* __restrict__ cls_b,
                       float* __restrict__ hb, float* __restrict__ tb) {
  int tid = blockIdx.x * blockDim.x + threadIdx.x;
  if (tid >= 2 * MM * OO) return;
  int which = (tid >= MM * OO);
  int r = which ? tid - MM * OO : tid;
  int m = r / OO, o = r % OO;
  const _Float16* row = (which ? t16 : h16) + (size_t)m * BSP;
  const float* w = cls_w + (size_t)o * HSZ + (which ? (BS + 1) : 0);
  float acc = w[BS];                                // "ones" column weight
  for (int i = 0; i < BS; ++i) acc += (float)row[i] * w[i];
  if (!which) acc += cls_b[o];
  (which ? tb : hb)[r] = acc;
}

// ---------------- GEMM2: hU[m][o*128+j] = sum_i h[m][i] * U[o][i][j] ----------------
__global__ __launch_bounds__(128) void k_hu(const _Float16* __restrict__ h16,
                                            const _Float16* __restrict__ ut,
                                            _Float16* __restrict__ hu) {
  int lane = threadIdx.x & 31, wave = threadIdx.x >> 5;
  int mt = blockIdx.x;
  int n0 = (blockIdx.y * 4 + wave) * 16;           // 80 N-tiles total
  const _Float16* A  = h16 + (size_t)mt * 16 * BSP;
  const _Float16* Bt = ut + (size_t)n0 * BSP;
  v8f c = {};
#pragma unroll
  for (int kk = 0; kk < BSP; kk += 32) {
    v16h a = frag_a_f16(A + kk, BSP, lane);
    v16h b = frag_b_f16(Bt + kk, BSP, lane);
    c = wmma32(a, b, c);
  }
  int g = lane >> 4, nl = lane & 15;
#pragma unroll
  for (int v = 0; v < 8; ++v) {
    int m = mt * 16 + v + 8 * g;
    hu[(size_t)m * NU + n0 + nl] = (_Float16)c[v];
  }
}

// ---------------- scores: per (b,x) block, out[y,o] = t · hUx^T, fused epilogue ----
// M = y (255, 16 tiles), N = o (10 padded to 16), K = 128.
// B operand (hUx^T, 16x128 f16 = 4KB) staged in LDS once, shared by all 4 waves.
// Each half-wave stores 10 contiguous floats per (x,y); a block writes one
// contiguous 10.2KB (y,o) slab of the output.
__global__ __launch_bounds__(128) void k_scores(const _Float16* __restrict__ hu,
                                                const _Float16* __restrict__ t16,
                                                const float* __restrict__ hb,
                                                const float* __restrict__ tb,
                                                const float* __restrict__ wtab,
                                                float* __restrict__ out) {
  __shared__ _Float16 bsh[16 * BSP];                // 4KB: rows = o (zero-padded 10->16)
  int lane = threadIdx.x & 31, wave = threadIdx.x >> 5;
  int x = blockIdx.x;                               // 0..254
  int b = blockIdx.y;                               // 0..31

  // stage hUx row (1280 halves) into LDS, zero-pad to 2048 halves
  const _Float16* hurow = hu + (size_t)(b * SS + x) * NU;
  for (int i = threadIdx.x; i < 1024; i += 128) {   // 1024 dwords = 2048 halves
    unsigned w = 0;
    if (2 * i < NU) w = *(const unsigned*)(hurow + 2 * i);
    *(unsigned*)(&bsh[2 * i]) = w;
  }
  __syncthreads();

  const _Float16* At = t16 + (size_t)b * SS * BSP;  // rows = y
  int ybase = wave * 64;                            // wave covers 4 y-tiles
  v8f c0 = {}, c1 = {}, c2 = {}, c3 = {};
#pragma unroll
  for (int kk = 0; kk < BSP; kk += 32) {
    v16h bf = frag_b_f16(bsh + kk, BSP, lane);      // ds_load from LDS
    v16h a0 = frag_a_f16(At + (size_t)(ybase     ) * BSP + kk, BSP, lane);
    v16h a1 = frag_a_f16(At + (size_t)(ybase + 16) * BSP + kk, BSP, lane);
    v16h a2 = frag_a_f16(At + (size_t)(ybase + 32) * BSP + kk, BSP, lane);
    v16h a3 = frag_a_f16(At + (size_t)(ybase + 48) * BSP + kk, BSP, lane);
    c0 = wmma32(a0, bf, c0);
    c1 = wmma32(a1, bf, c1);
    c2 = wmma32(a2, bf, c2);
    c3 = wmma32(a3, bf, c3);
  }

  int g = lane >> 4, nl = lane & 15;                // nl = o
  if (nl >= OO) return;                             // inactive N-pad lanes
  float hbv = hb[((size_t)b * SS + x) * OO + nl];
  float* outx = out + ((size_t)b * SS + x) * SS * OO;
#pragma unroll
  for (int t = 0; t < 4; ++t) {
    v8f cc = (t == 0) ? c0 : (t == 1) ? c1 : (t == 2) ? c2 : c3;
#pragma unroll
    for (int v = 0; v < 8; ++v) {
      int y = ybase + t * 16 + v + 8 * g;
      if (y < SS) {
        int d = y - x + 1; d = (d > 0) ? d : 0;     // width index, 0..255
        float val = cc[v] + hbv + tb[((size_t)b * SS + y) * OO + nl] + wtab[d * OO + nl];
        outx[(size_t)y * OO + nl] = val;
      }
    }
  }
}

// ---------------- host ----------------
extern "C" void kernel_launch(void* const* d_in, const int* in_sizes, int n_in,
                              void* d_out, int out_size, void* d_ws, size_t ws_size,
                              hipStream_t stream) {
  (void)in_sizes; (void)n_in; (void)out_size; (void)ws_size;
  const float* state  = (const float*)d_in[0];
  const float* head_w = (const float*)d_in[1];
  const float* head_b = (const float*)d_in[2];
  const float* tail_w = (const float*)d_in[3];
  const float* tail_b = (const float*)d_in[4];
  const float* U      = (const float*)d_in[5];
  const float* wtable = (const float*)d_in[6];
  const float* cls_w  = (const float*)d_in[7];
  const float* cls_b  = (const float*)d_in[8];
  float* out = (float*)d_out;

  char* ws = (char*)d_ws;
  size_t off = 0;
  auto alloc = [&](size_t bytes) -> void* {
    void* p = ws + off;
    off += (bytes + 255) & ~(size_t)255;
    return p;
  };
  _Float16* hwt = (_Float16*)alloc((size_t)BSP * HH * 2);
  _Float16* twt = (_Float16*)alloc((size_t)BSP * HH * 2);
  _Float16* ut  = (_Float16*)alloc((size_t)NU * BSP * 2);
  _Float16* h16 = (_Float16*)alloc((size_t)(MM + 16) * BSP * 2);   // +16 pad rows
  _Float16* t16 = (_Float16*)alloc((size_t)(MM + 16) * BSP * 2);
  _Float16* hu  = (_Float16*)alloc((size_t)(MM + 16) * NU * 2);
  float* hb   = (float*)alloc((size_t)MM * OO * 4);
  float* tb   = (float*)alloc((size_t)MM * OO * 4);
  float* wtab = (float*)alloc((size_t)256 * OO * 4);

  k_pack_w   <<<(BSP * HH + 255) / 256, 256, 0, stream>>>(head_w, tail_w, hwt, twt);
  k_pack_u   <<<(NU * BSP + 255) / 256, 256, 0, stream>>>(U, ut);
  k_pack_wtab<<<(256 * OO + 255) / 256, 256, 0, stream>>>(wtable, cls_w, wtab);

  k_proj<<<MT, 128, 0, stream>>>(state, hwt, head_b, h16);
  k_proj<<<MT, 128, 0, stream>>>(state, twt, tail_b, t16);

  k_bias<<<(2 * MM * OO + 255) / 256, 256, 0, stream>>>(h16, t16, cls_w, cls_b, hb, tb);

  k_hu<<<dim3(MT, 20), 128, 0, stream>>>(h16, ut, hu);

  k_scores<<<dim3(SS, BB), 128, 0, stream>>>(hu, t16, hb, tb, wtab, out);
}
